// GraphConvolution_68367289418480
// MI455X (gfx1250) — matrix operand order repaired and checked by reading, not compiled, not run on hardware
//
#include <hip/hip_runtime.h>

// ---- WMMA vector types (gfx1250, wave32) ----
typedef __attribute__((ext_vector_type(16))) __bf16 v16bf;
typedef __attribute__((ext_vector_type(8)))  float  v8f;

union FragA {
    uint4 q[2];
    v16bf v;
};

#define BM 128
#define BN 128
#define BK 32
#define LDT 48                    // padded LDS row stride (96 B: 32B-aligned rows)
#define TILE_BYTES (BM * LDT * 2) // 12288 B per matrix tile
#define BUF_BYTES (2 * TILE_BYTES)// 24576 B per double-buffer slot (A + B)
#define SMEM_BYTES (2 * BUF_BYTES)// 49152 B total dynamic LDS

// ---------------------------------------------------------------------------
// CDNA5 async global->LDS DMA (tracked by ASYNCcnt; no VGPR staging)
// vdst = per-lane 32-bit LDS byte address, vaddr = per-lane 64-bit global addr
// ---------------------------------------------------------------------------
__device__ __forceinline__ void async_copy16(uint32_t lds_off, const void* g) {
    asm volatile("global_load_async_to_lds_b128 %0, %1, off"
                 :: "v"(lds_off), "v"(g)
                 : "memory");
}
__device__ __forceinline__ void wait_async0() {
    asm volatile("s_wait_asynccnt 0x0" ::: "memory");
}

// ---------------------------------------------------------------------------
// fp32 -> bf16 convert (grid-stride)
// ---------------------------------------------------------------------------
__global__ void cvt_bf16_kernel(const float* __restrict__ in,
                                __bf16* __restrict__ out, size_t n) {
    size_t i = (size_t)blockIdx.x * blockDim.x + threadIdx.x;
    size_t stride = (size_t)gridDim.x * blockDim.x;
    for (; i < n; i += stride) out[i] = (__bf16)in[i];
}

// ---------------------------------------------------------------------------
// fp32 [rows][cols] -> bf16 transposed [cols][rows]
// ---------------------------------------------------------------------------
__global__ __launch_bounds__(256) void transpose_cvt_kernel(
    const float* __restrict__ in, __bf16* __restrict__ out,
    int rows, int cols) {
    __shared__ float tile[32][33];
    const int c0 = blockIdx.x * 32;
    const int r0 = blockIdx.y * 32;
    const int tx = threadIdx.x;       // 0..31
    const int ty = threadIdx.y;       // 0..7
    for (int i = 0; i < 32; i += 8)
        tile[ty + i][tx] = in[(size_t)(r0 + ty + i) * cols + c0 + tx];
    __syncthreads();
    for (int i = 0; i < 32; i += 8)
        out[(size_t)(c0 + ty + i) * rows + r0 + tx] = (__bf16)tile[tx][ty + i];
}

// ---------------------------------------------------------------------------
// bf16 WMMA GEMM:  C[M][N] = A[M][K] * Bt[N][K]^T   (+bias, ReLU optional)
// 256 threads (8 wave32), 128x128 block tile, K-tile 32, double-buffered LDS
// fed by global_load_async_to_lds_b128 under the WMMA shadow.
// Waves: 2 along M (64 rows) x 4 along N (32 cols); each wave 4x2 WMMA tiles.
// ---------------------------------------------------------------------------
template <bool BIAS_RELU, typename OutT>
__global__ __launch_bounds__(256) void wmma_gemm_bf16(
    const __bf16* __restrict__ A,    // [M][K] row-major
    const __bf16* __restrict__ Bt,   // [N][K] row-major (i.e. B transposed)
    const float*  __restrict__ bias, // [N] or nullptr
    OutT* __restrict__ C,            // [M][N]
    int M, int N, int K) {
    extern __shared__ __align__(16) char smem[];

    const int tid  = threadIdx.x;
    const int lane = tid & 31;
    const int wave = tid >> 5;
    const int wm = (wave & 1) * 64;   // wave M offset in block tile
    const int wn = (wave >> 1) * 32;  // wave N offset in block tile
    const int hi = lane >> 4;         // lane half: 0 or 1
    const int lm = lane & 15;

    const int m0 = blockIdx.y * BM;
    const int n0 = blockIdx.x * BN;

    // cooperative tile copy: 256 threads x 16B per chunk, 2 chunks per matrix
    const int row = tid >> 2;         // 0..63
    const int cg  = (tid & 3) * 8;    // bf16 col offset: 0,8,16,24

    // flat-LDS addr[31:0] == wave-relative LDS byte address (ISA 10.2)
    const uint32_t lds_base = (uint32_t)(size_t)smem;
    const uint32_t offA0 = (uint32_t)((row * LDT + cg) * 2);
    const uint32_t offA1 = (uint32_t)(((row + 64) * LDT + cg) * 2);

    const __bf16* gA0 = A  + (size_t)(m0 + row) * K + cg;
    const __bf16* gA1 = A  + (size_t)(m0 + 64 + row) * K + cg;
    const __bf16* gB0 = Bt + (size_t)(n0 + row) * K + cg;
    const __bf16* gB1 = Bt + (size_t)(n0 + 64 + row) * K + cg;

    v8f acc[4][2] = {};               // 8 accumulators * 8 VGPRs

    // prologue: DMA tile 0 into buffer 0
    {
        const uint32_t bb = lds_base;
        async_copy16(bb + offA0, gA0);
        async_copy16(bb + offA1, gA1);
        async_copy16(bb + TILE_BYTES + offA0, gB0);
        async_copy16(bb + TILE_BYTES + offA1, gB1);
        gA0 += BK; gA1 += BK; gB0 += BK; gB1 += BK;
    }

    const int steps = K / BK;
    for (int s = 0; s < steps; ++s) {
        const int cur = s & 1;

        wait_async0();        // my DMA for tile s has landed
        __syncthreads();      // everyone's tile-s DMA landed; tile s-1 reads done

        if (s + 1 < steps) {  // DMA tile s+1 into the other buffer (overlaps WMMA)
            const uint32_t bb = lds_base + (uint32_t)((cur ^ 1) * BUF_BYTES);
            async_copy16(bb + offA0, gA0);
            async_copy16(bb + offA1, gA1);
            async_copy16(bb + TILE_BYTES + offA0, gB0);
            async_copy16(bb + TILE_BYTES + offA1, gB1);
            gA0 += BK; gA1 += BK; gB0 += BK; gB1 += BK;
            if (s + 2 < steps) {  // warm L2 two tiles ahead
                __builtin_prefetch(gA0, 0, 1);
                __builtin_prefetch(gB0, 0, 1);
            }
        }

        const __bf16* sA = (const __bf16*)(smem + cur * BUF_BYTES);
        const __bf16* sB = (const __bf16*)(smem + cur * BUF_BYTES + TILE_BYTES);

        // A 16x32 bf16 layout: lane half hi holds 16 contiguous K at 16*hi, row lm
        FragA a[4];
#pragma unroll
        for (int i = 0; i < 4; i++) {
            const __bf16* base = sA + (wm + i * 16 + lm) * LDT;
            a[i].q[0] = *(const uint4*)(base + 16 * hi);
            a[i].q[1] = *(const uint4*)(base + 16 * hi + 8);
        }
        // B 32x16 bf16 layout: lane half hi holds 16 contiguous K at 16*hi, column lm
        v16bf b[2];
#pragma unroll
        for (int j = 0; j < 2; j++)
            b[j] = *(const v16bf*)(sB + (wn + j * 16 + lm) * LDT + 16 * hi);

#pragma unroll
        for (int i = 0; i < 4; i++)
#pragma unroll
            for (int j = 0; j < 2; j++)
                acc[i][j] = __builtin_amdgcn_wmma_f32_16x16x32_bf16(
                    false, a[i].v, false, b[j], (short)0, acc[i][j], false, false);

        __syncthreads();      // tile-s reads done before next iter DMAs over it
    }

    // ---- epilogue: C/D layout VGPR r -> M = r + 8*hi, N = lane&15 ----
    const int cm8 = hi * 8;
#pragma unroll
    for (int i = 0; i < 4; i++) {
#pragma unroll
        for (int j = 0; j < 2; j++) {
            const int gmb = m0 + wm + i * 16 + cm8;
            const int gn  = n0 + wn + j * 16 + lm;
            float bv = 0.0f;
            if constexpr (BIAS_RELU) bv = bias[gn];
#pragma unroll
            for (int r = 0; r < 8; r++) {
                float v = acc[i][j][r];
                if constexpr (BIAS_RELU) {
                    v += bv;
                    v = v > 0.0f ? v : 0.0f;
                }
                C[(size_t)(gmb + r) * N + gn] = (OutT)v;
            }
        }
    }
}

// ---------------------------------------------------------------------------
// launch
// ---------------------------------------------------------------------------
extern "C" void kernel_launch(void* const* d_in, const int* in_sizes, int n_in,
                              void* d_out, int out_size, void* d_ws, size_t ws_size,
                              hipStream_t stream) {
    constexpr int Nn   = 8192;
    constexpr int Din  = 1024;
    constexpr int Dout = 1024;

    const float* features = (const float*)d_in[0]; // [N][Din]
    const float* adj      = (const float*)d_in[1]; // [N][N]
    const float* W        = (const float*)d_in[2]; // [Dout][Din]
    const float* bvec     = (const float*)d_in[3]; // [Dout]
    float* out = (float*)d_out;                    // [N][Dout]

    // workspace layout (bf16 staging)
    char* ws = (char*)d_ws;
    __bf16* adjB = (__bf16*)ws;                                 // 128 MiB
    ws += (size_t)Nn * Nn * sizeof(__bf16);
    __bf16* fT   = (__bf16*)ws;                                 // [Din][N] 16 MiB
    ws += (size_t)Din * Nn * sizeof(__bf16);
    __bf16* Wb   = (__bf16*)ws;                                 // [Dout][Din] 2 MiB
    ws += (size_t)Dout * Din * sizeof(__bf16);
    __bf16* aggB = (__bf16*)ws;                                 // [N][Din] 16 MiB

    // 1) adj -> bf16
    cvt_bf16_kernel<<<8192, 256, 0, stream>>>(adj, adjB, (size_t)Nn * Nn);
    // 2) features -> bf16 transposed: fT[d][j] = features[j][d]
    transpose_cvt_kernel<<<dim3(Din / 32, Nn / 32), dim3(32, 8), 0, stream>>>(
        features, fT, Nn, Din);
    // 3) W -> bf16 (row-major [o][d] is already the Bt layout for GEMM-2)
    cvt_bf16_kernel<<<2048, 256, 0, stream>>>(W, Wb, (size_t)Dout * Din);

    // 4) agg = adj @ features   (M=8192, N=1024, K=8192), bf16 out
    wmma_gemm_bf16<false, __bf16>
        <<<dim3(Din / BN, Nn / BM), 256, SMEM_BYTES, stream>>>(
            adjB, fT, nullptr, aggB, Nn, Din, Nn);

    // 5) out = relu(agg @ W^T + b)  (M=8192, N=1024, K=1024), f32 out
    wmma_gemm_bf16<true, float>
        <<<dim3(Dout / BN, Nn / BM), 256, SMEM_BYTES, stream>>>(
            aggB, Wb, bvec, out, Nn, Dout, Din);
}
